// MultiModalBEVENet_16604343566430
// MI455X (gfx1250) — matrix operand order repaired and checked by reading, not compiled, not run on hardware
//
#include <hip/hip_runtime.h>

// ---------------------------------------------------------------------------
// MI455X (gfx1250) implementation of MultiModalBEVENet forward pass.
// - Every conv & every attention matmul runs on v_wmma_f32_16x16x32_bf16.
// - Conv weights stream through the Tensor Data Mover (tensor_load_to_lds,
//   s_wait_tensorcnt) into LDS; im2col tiles are built per-lane (float4 fast
//   path for 1x1 convs).
// - Attention is fused flash-style: the 6400x6400 score matrix is never
//   materialized.
// ---------------------------------------------------------------------------

typedef __attribute__((ext_vector_type(16))) __bf16 v16bf;
typedef __attribute__((ext_vector_type(8)))  float  v8f;
typedef unsigned int u32x4 __attribute__((ext_vector_type(4)));
typedef int          i32x4 __attribute__((ext_vector_type(4)));
typedef int          i32x8 __attribute__((ext_vector_type(8)));

__device__ __forceinline__ __bf16 f2bf(float f) {
    union { float f; unsigned int u; } a; a.f = f;
    unsigned int r = a.u + 0x7fffu + ((a.u >> 16) & 1u);   // RNE
    union { unsigned short u; __bf16 h; } c; c.u = (unsigned short)(r >> 16);
    return c.h;
}

__device__ __forceinline__ float silu_f(float x) {
    return x / (1.0f + __expf(-x));
}

// Issue one TDM 2-D tile load: tile_dim0=32 cols x tile_dim1=16 rows of f32,
// row stride = rowStride elements, remaining-extent zero-fill via tensor dims.
__device__ __forceinline__ void tdm_load_tile(const float* gsrc, unsigned ldsOff,
                                              unsigned remCols, unsigned remRows,
                                              unsigned rowStride) {
    unsigned long long ga = (unsigned long long)gsrc;
    u32x4 g0 = { 1u,                                   // count=1, user mode
                 ldsOff,                               // lds_addr (bytes)
                 (unsigned)ga,                         // global_addr[31:0]
                 ((unsigned)(ga >> 32) & 0x01FFFFFFu)  // global_addr[56:32]
                 | 0x80000000u };                      // type=2 ("image")
    unsigned long long st0 = (unsigned long long)rowStride;
    i32x8 g1 = { (int)(2u << 16),                                  // data_size=4B
                 (int)((remCols & 0xFFFFu) << 16),                 // tensor_dim0 lo
                 (int)(((remCols >> 16) & 0xFFFFu) |
                       ((remRows & 0xFFFFu) << 16)),               // td0 hi | td1 lo
                 (int)(((remRows >> 16) & 0xFFFFu) | (32u << 16)), // td1 hi | tile0=32
                 (int)16,                                          // tile_dim1=16
                 (int)(st0 & 0xFFFFFFFFu),                         // dim0 stride lo
                 (int)((st0 >> 32) & 0xFFFFu),                     // dim0 stride hi
                 0 };
    i32x4 gz = { 0, 0, 0, 0 };
#if __clang_major__ >= 23
    i32x8 gz8 = { 0, 0, 0, 0, 0, 0, 0, 0 };
    __builtin_amdgcn_tensor_load_to_lds(g0, g1, gz, gz, gz8, 0);
#else
    __builtin_amdgcn_tensor_load_to_lds(g0, g1, gz, gz, 0);
#endif
}

// ---------------------------------------------------------------------------
// Implicit-GEMM conv (NCHW, groups==1) with WMMA bf16.
// One wave per block computes 16 output channels x 64 output pixels:
// the A (weights) fragment is reused across 4 B fragments -> 4 WMMAs/chunk.
// Weights are staged via the Tensor Data Mover; im2col rows per-lane (kk=tid).
// Epilogue: y = S[c]*acc + B[c]; optional SiLU; optional residual add.
// Channel offsets allow reading slices and writing into concat buffers.
// ---------------------------------------------------------------------------
__global__ __launch_bounds__(32)
void conv_wmma(const float* __restrict__ in, const float* __restrict__ Wt,
               const float* __restrict__ Sc, const float* __restrict__ Bi,
               const float* __restrict__ Res, float* __restrict__ out,
               int Cin, int Cout, int H, int Wd, int OH, int OW,
               int KH, int KW, int stride, int pad,
               int CinTot, int icoff, int CoutTot, int ocoff,
               int ResCt, int resoff, int act)
{
    const int tid  = threadIdx.x;
    const int half = tid >> 4;
    const int lane = tid & 15;
    const int OHW  = OH * OW;
    const int n0   = blockIdx.x * 64;
    const int m0   = blockIdx.y * 16;
    const int bi   = blockIdx.z;
    const int KHW  = KH * KW;
    const int Ktot = Cin * KHW;
    const bool onexone = (KH == 1 && KW == 1 && stride == 1 && pad == 0);

    __shared__ float  At[16][32];   // [m][k]   (fp32, TDM-filled)
    __shared__ __bf16 Bt[32][64];   // [k][n]

    const unsigned atOff = (unsigned)(unsigned long long)(&At[0][0]);

    __builtin_prefetch(Wt + (size_t)m0 * Ktot, 0, 3);   // global_prefetch_b8

    v8f acc0 = {}, acc1 = {}, acc2 = {}, acc3 = {};

    for (int k0 = 0; k0 < Ktot; k0 += 32) {
        // ---- stage A (weights, OIHW flat == M x K) via Tensor Data Mover ----
        tdm_load_tile(Wt + (size_t)m0 * Ktot + k0, atOff,
                      (unsigned)(Ktot - k0), (unsigned)(Cout - m0),
                      (unsigned)Ktot);

        // ---- stage B (im2col inputs, K x N): lane owns row kk = tid ----
        {
            const int kk = tid;
            const int k  = k0 + kk;
            const bool kvalid = (k < Ktot);
            int ci = 0, kh = 0, kw = 0;
            if (kvalid) {
                ci = k / KHW; int r = k - ci * KHW;
                kh = r / KW;  kw = r - kh * KW;
            }
            const float* src =
                in + (size_t)(bi * CinTot + icoff + ci) * H * Wd;
            if (onexone) {
                if (kvalid && n0 + 64 <= OHW) {
                    const float4* p4 = (const float4*)(src + n0);
                    #pragma unroll
                    for (int t = 0; t < 16; ++t) {
                        float4 v = p4[t];
                        Bt[kk][t * 4 + 0] = f2bf(v.x);
                        Bt[kk][t * 4 + 1] = f2bf(v.y);
                        Bt[kk][t * 4 + 2] = f2bf(v.z);
                        Bt[kk][t * 4 + 3] = f2bf(v.w);
                    }
                } else {
                    for (int j = 0; j < 64; ++j) {
                        int sp = n0 + j;
                        float v = (kvalid && sp < OHW) ? src[sp] : 0.0f;
                        Bt[kk][j] = f2bf(v);
                    }
                }
            } else {
                int oh = n0 / OW, ow = n0 - oh * OW;
                const int ihb = kh - pad, iwb = kw - pad;
                for (int j = 0; j < 64; ++j) {
                    float v = 0.0f;
                    if (kvalid && n0 + j < OHW) {
                        int ih = oh * stride + ihb;
                        int iw = ow * stride + iwb;
                        if (ih >= 0 && ih < H && iw >= 0 && iw < Wd)
                            v = src[(size_t)ih * Wd + iw];
                    }
                    Bt[kk][j] = f2bf(v);
                    if (++ow == OW) { ow = 0; ++oh; }
                }
            }
        }

        __builtin_amdgcn_s_wait_tensorcnt(0);
        __syncthreads();

        // ---- fragments + 4 WMMAs (A reused) ----
        v16bf a;
        #pragma unroll
        for (int i = 0; i < 16; ++i) {
            int grp = i >> 3, wi = i & 7;
            int ka = wi + 8 * half + 16 * grp;   // A-matrix 16x32 bf16 lane layout
            a[i] = f2bf(At[lane][ka]);
        }
        v16bf b0, b1, b2, b3;
        #pragma unroll
        for (int i = 0; i < 16; ++i) {
            int kr = (i & 7) + 8 * half + 16 * (i >> 3);              // B-matrix 32x16 lane layout
            b0[i] = Bt[kr][lane];
            b1[i] = Bt[kr][16 + lane];
            b2[i] = Bt[kr][32 + lane];
            b3[i] = Bt[kr][48 + lane];
        }
        acc0 = __builtin_amdgcn_wmma_f32_16x16x32_bf16(false, a, false, b0, (short)0, acc0, false, false);
        acc1 = __builtin_amdgcn_wmma_f32_16x16x32_bf16(false, a, false, b1, (short)0, acc1, false, false);
        acc2 = __builtin_amdgcn_wmma_f32_16x16x32_bf16(false, a, false, b2, (short)0, acc2, false, false);
        acc3 = __builtin_amdgcn_wmma_f32_16x16x32_bf16(false, a, false, b3, (short)0, acc3, false, false);
        __syncthreads();
    }

    // ---- epilogue ----
    #pragma unroll
    for (int t = 0; t < 4; ++t) {
        const v8f* accp = (t == 0) ? &acc0 : (t == 1) ? &acc1 : (t == 2) ? &acc2 : &acc3;
        int sp = n0 + t * 16 + lane;
        if (sp < OHW) {
            #pragma unroll
            for (int r = 0; r < 8; ++r) {
                int m = m0 + r + 8 * half;          // C/D layout: row = r + 8*half
                if (m < Cout) {
                    float v = (*accp)[r];
                    v = (Sc ? Sc[m] : 1.0f) * v + (Bi ? Bi[m] : 0.0f);
                    if (act) v = silu_f(v);
                    if (Res) v += Res[((size_t)(bi * ResCt + resoff + m)) * OHW + sp];
                    out[((size_t)(bi * CoutTot + ocoff + m)) * OHW + sp] = v;
                }
            }
        }
    }
}

// ---------------------------------------------------------------------------
// Fused flash attention:  O[c,n] = sum_m V[c,m] * softmax_m(scale * Q[:,n].K[:,m])
// One wave per 16-query tile, streaming keys 32 at a time.
// KC = ceil(Cqk/32) score K-chunks, CT = Cv/16 output row tiles.
// ---------------------------------------------------------------------------
template <int KC, int CT>
__global__ __launch_bounds__(32)
void attn_fused(const float* __restrict__ Q, int q_off,
                const float* __restrict__ K, int k_off,
                const float* __restrict__ V, int v_off,
                float* __restrict__ O, int o_off,
                int Cqk, int N, int M, float scale)
{
    const int tid  = threadIdx.x;
    const int half = tid >> 4;
    const int lane = tid & 15;
    const int n0   = blockIdx.x * 16;
    const int nq   = n0 + lane;

    __shared__ __bf16 Pt[32][16];   // [m_local][n]
    __shared__ float  rstat[16];

    // Q^T tile: A row = query n, col = channel c
    v16bf aq[KC];
    #pragma unroll
    for (int cc = 0; cc < KC; ++cc) {
        v16bf a;
        #pragma unroll
        for (int i = 0; i < 16; ++i) {
            int grp = i >> 3, wi = i & 7;
            int c = cc * 32 + wi + 8 * half + 16 * grp;
            float v = (c < Cqk && nq < N) ? Q[(size_t)(q_off + c) * N + nq] : 0.0f;
            a[i] = f2bf(v);
        }
        aq[cc] = a;
    }

    v8f oacc[CT];
    #pragma unroll
    for (int t = 0; t < CT; ++t) oacc[t] = (v8f){};
    float mst[8], lst[8];
    #pragma unroll
    for (int r = 0; r < 8; ++r) { mst[r] = -3.0e38f; lst[r] = 0.0f; }

    for (int mb = 0; mb < M; mb += 32) {
        v8f s0 = (v8f){}, s1 = (v8f){};
        #pragma unroll
        for (int cc = 0; cc < KC; ++cc) {
            v16bf b0, b1;
            #pragma unroll
            for (int i = 0; i < 16; ++i) {
                int c = cc * 32 + (i & 7) + 8 * half + 16 * (i >> 3);
                float v0 = (c < Cqk) ? K[(size_t)(k_off + c) * M + mb + lane] : 0.0f;
                float v1 = (c < Cqk) ? K[(size_t)(k_off + c) * M + mb + 16 + lane] : 0.0f;
                b0[i] = f2bf(v0);
                b1[i] = f2bf(v1);
            }
            s0 = __builtin_amdgcn_wmma_f32_16x16x32_bf16(false, aq[cc], false, b0,
                                                         (short)0, s0, false, false);
            s1 = __builtin_amdgcn_wmma_f32_16x16x32_bf16(false, aq[cc], false, b1,
                                                         (short)0, s1, false, false);
        }
        // online softmax over the 32 keys of this step (row = query)
        float corr[8], p0[8], p1[8];
        #pragma unroll
        for (int r = 0; r < 8; ++r) {
            float a0 = s0[r] * scale, a1 = s1[r] * scale;
            float tm = fmaxf(a0, a1);
            tm = fmaxf(tm, __shfl_xor(tm, 1, 32));
            tm = fmaxf(tm, __shfl_xor(tm, 2, 32));
            tm = fmaxf(tm, __shfl_xor(tm, 4, 32));
            tm = fmaxf(tm, __shfl_xor(tm, 8, 32));
            float mn = fmaxf(mst[r], tm);
            float cr = __expf(mst[r] - mn);
            float e0 = __expf(a0 - mn), e1 = __expf(a1 - mn);
            float ts = e0 + e1;
            ts += __shfl_xor(ts, 1, 32);
            ts += __shfl_xor(ts, 2, 32);
            ts += __shfl_xor(ts, 4, 32);
            ts += __shfl_xor(ts, 8, 32);
            lst[r] = lst[r] * cr + ts;
            mst[r] = mn;
            corr[r] = cr; p0[r] = e0; p1[r] = e1;
        }
        if (lane == 0) {
            #pragma unroll
            for (int r = 0; r < 8; ++r) rstat[r + 8 * half] = corr[r];
        }
        __syncthreads();
        float cn = rstat[lane];
        #pragma unroll
        for (int t = 0; t < CT; ++t)
            #pragma unroll
            for (int r = 0; r < 8; ++r) oacc[t][r] *= cn;

        // transpose P through LDS into B-matrix layout for the O GEMM
        #pragma unroll
        for (int r = 0; r < 8; ++r) {
            Pt[lane][r + 8 * half]      = f2bf(p0[r]);
            Pt[lane + 16][r + 8 * half] = f2bf(p1[r]);
        }
        __syncthreads();
        v16bf bp;
        #pragma unroll
        for (int i = 0; i < 16; ++i) bp[i] = Pt[(i & 7) + 8 * half + 16 * (i >> 3)][lane];
        #pragma unroll
        for (int t = 0; t < CT; ++t) {
            v16bf av;
            #pragma unroll
            for (int i = 0; i < 16; ++i) {
                int grp = i >> 3, wi = i & 7;
                int kk = wi + 8 * half + 16 * grp;
                av[i] = f2bf(V[(size_t)(v_off + t * 16 + lane) * M + mb + kk]);
            }
            oacc[t] = __builtin_amdgcn_wmma_f32_16x16x32_bf16(false, av, false, bp,
                                                              (short)0, oacc[t], false, false);
        }
        __syncthreads();
    }

    if (lane == 0) {
        #pragma unroll
        for (int r = 0; r < 8; ++r) rstat[r + 8 * half] = 1.0f / lst[r];
    }
    __syncthreads();
    float inv = rstat[lane];
    #pragma unroll
    for (int t = 0; t < CT; ++t) {
        #pragma unroll
        for (int r = 0; r < 8; ++r) {
            int c = t * 16 + r + 8 * half;
            O[(size_t)(o_off + c) * N + n0 + lane] = oacc[t][r] * inv;
        }
    }
}

// ---------------------------------------------------------------------------
// Elementwise / small helper kernels (B folded where always 1)
// ---------------------------------------------------------------------------
__global__ void fill_zero(float* p, int n) {
    int i = blockIdx.x * blockDim.x + threadIdx.x;
    if (i < n) p[i] = 0.0f;
}

__global__ void softmax_ch(float* buf, int C, int HW, int Bn) {
    int idx = blockIdx.x * blockDim.x + threadIdx.x;
    if (idx >= Bn * HW) return;
    int b = idx / HW, p = idx - b * HW;
    float* base = buf + (size_t)b * C * HW + p;
    float m = -3.0e38f;
    for (int c = 0; c < C; ++c) m = fmaxf(m, base[(size_t)c * HW]);
    float s = 0.0f;
    for (int c = 0; c < C; ++c) s += __expf(base[(size_t)c * HW] - m);
    float inv = 1.0f / s;
    for (int c = 0; c < C; ++c)
        base[(size_t)c * HW] = __expf(base[(size_t)c * HW] - m) * inv;
}

__global__ void maxpool5(const float* __restrict__ in, float* __restrict__ out,
                         int C, int H, int Wd, int icoff, int ocoff) {
    int idx = blockIdx.x * blockDim.x + threadIdx.x;
    if (idx >= C * H * Wd) return;
    int c = idx / (H * Wd); int r = idx - c * H * Wd;
    int y = r / Wd, x = r - y * Wd;
    float m = -3.0e38f;
    for (int dy = -2; dy <= 2; ++dy)
        for (int dx = -2; dx <= 2; ++dx) {
            int yy = y + dy, xx = x + dx;
            if (yy >= 0 && yy < H && xx >= 0 && xx < Wd)
                m = fmaxf(m, in[((size_t)(icoff + c) * H + yy) * Wd + xx]);
        }
    out[((size_t)(ocoff + c) * H + y) * Wd + x] = m;
}

__global__ void dwconv3(const float* __restrict__ in, const float* __restrict__ Wt,
                        const float* __restrict__ Sc, const float* __restrict__ Bi,
                        const float* __restrict__ Res, float* __restrict__ out,
                        int C, int H, int Wd, int icoff, int ocoff, int resoff, int act) {
    int idx = blockIdx.x * blockDim.x + threadIdx.x;
    if (idx >= C * H * Wd) return;
    int c = idx / (H * Wd); int r = idx - c * H * Wd;
    int y = r / Wd, x = r - y * Wd;
    float acc = 0.0f;
    for (int kh = 0; kh < 3; ++kh)
        for (int kw = 0; kw < 3; ++kw) {
            int ih = y - 1 + kh, iw = x - 1 + kw;
            if (ih >= 0 && ih < H && iw >= 0 && iw < Wd)
                acc += in[((size_t)(icoff + c) * H + ih) * Wd + iw] * Wt[c * 9 + kh * 3 + kw];
        }
    acc = Sc[c] * acc + Bi[c];
    if (act) acc = silu_f(acc);
    if (Res) acc += Res[((size_t)(resoff + c) * H + y) * Wd + x];
    out[((size_t)(ocoff + c) * H + y) * Wd + x] = acc;
}

// ---------------------------------------------------------------------------
// Geometry: per-camera transform matrices (inv(post_rots), rots*inv(intrins))
// ---------------------------------------------------------------------------
__device__ void inv3(const float* a, float* o) {
    float d = a[0] * (a[4] * a[8] - a[5] * a[7])
            - a[1] * (a[3] * a[8] - a[5] * a[6])
            + a[2] * (a[3] * a[7] - a[4] * a[6]);
    float id = 1.0f / d;
    o[0] = (a[4] * a[8] - a[5] * a[7]) * id; o[1] = (a[2] * a[7] - a[1] * a[8]) * id;
    o[2] = (a[1] * a[5] - a[2] * a[4]) * id; o[3] = (a[5] * a[6] - a[3] * a[8]) * id;
    o[4] = (a[0] * a[8] - a[2] * a[6]) * id; o[5] = (a[2] * a[3] - a[0] * a[5]) * id;
    o[6] = (a[3] * a[7] - a[4] * a[6]) * id; o[7] = (a[1] * a[6] - a[0] * a[7]) * id;
    o[8] = (a[0] * a[4] - a[1] * a[3]) * id;
}

__global__ void geom_prep(const float* rots, const float* trans, const float* intr,
                          const float* prots, const float* ptrans, float* gp) {
    int n = threadIdx.x;
    if (n >= 6) return;
    float iPR[9], iIn[9], C2E[9];
    inv3(prots + n * 9, iPR);
    inv3(intr + n * 9, iIn);
    const float* R = rots + n * 9;
    for (int i = 0; i < 3; ++i)
        for (int j = 0; j < 3; ++j) {
            float s = 0.0f;
            for (int k = 0; k < 3; ++k) s += R[i * 3 + k] * iIn[k * 3 + j];
            C2E[i * 3 + j] = s;
        }
    float* o = gp + n * 24;
    for (int i = 0; i < 9; ++i) { o[i] = iPR[i]; o[9 + i] = C2E[i]; }
    for (int i = 0; i < 3; ++i) { o[18 + i] = trans[n * 3 + i]; o[21 + i] = ptrans[n * 3 + i]; }
}

// Fused lift (x * depth) + voxel pooling (never materializes the lifted tensor)
__global__ void lift_voxel(const float* __restrict__ x, const float* __restrict__ dp,
                           const float* __restrict__ gp, float* __restrict__ vol) {
    int idx = blockIdx.x * blockDim.x + threadIdx.x;
    const int PTS = 6 * 41 * 8 * 22;
    if (idx >= PTS) return;
    int w = idx % 22; int t = idx / 22;
    int h = t % 8;    t /= 8;
    int dd = t % 41;  int n = t / 41;
    const float* g = gp + n * 24;
    float fx = (float)w * (351.0f / 21.0f);
    float fy = (float)h * (127.0f / 7.0f);
    float fd = 4.0f + (float)dd;
    float px = fx - g[21], py = fy - g[22], pz = fd - g[23];
    float qx = g[0] * px + g[1] * py + g[2] * pz;
    float qy = g[3] * px + g[4] * py + g[5] * pz;
    float qz = g[6] * px + g[7] * py + g[8] * pz;
    float rx = qx * qz, ry = qy * qz, rz = qz;
    float gx = g[9]  * rx + g[10] * ry + g[11] * rz + g[18];
    float gy = g[12] * rx + g[13] * ry + g[14] * rz + g[19];
    float gz = g[15] * rx + g[16] * ry + g[17] * rz + g[20];
    int ix = (int)((gx + 50.0f) / 1.25f);   // trunc toward zero == astype(int32)
    int iy = (int)((gy + 50.0f) / 1.25f);
    int iz = (int)((gz + 10.0f) / 20.0f);
    if (ix >= 0 && ix < 80 && iy >= 0 && iy < 80 && iz >= 0 && iz < 1) {
        float dw = dp[((size_t)(n * 41 + dd) * 8 + h) * 22 + w];
        const float* xf = x + (size_t)n * 64 * 176 + h * 22 + w;
        float* cell = vol + (size_t)ix * 80 + iy;
        for (int c = 0; c < 64; ++c)
            atomicAdd(cell + (size_t)c * 6400, xf[(size_t)c * 176] * dw);
    }
}

// ---------------------------------------------------------------------------
// Host orchestration
// ---------------------------------------------------------------------------
extern "C" void kernel_launch(void* const* d_in, const int* in_sizes, int n_in,
                              void* d_out, int out_size, void* d_ws, size_t ws_size,
                              hipStream_t stream)
{
    (void)in_sizes; (void)n_in; (void)out_size; (void)ws_size;

    const float* cam    = (const float*)d_in[0];
    const float* rots   = (const float*)d_in[1];
    const float* trans  = (const float*)d_in[2];
    const float* intr   = (const float*)d_in[3];
    const float* prots  = (const float*)d_in[4];
    const float* ptrans = (const float*)d_in[5];
    const float* lidar  = (const float*)d_in[6];

    struct CP { const float *w, *s, *b; };
    struct PC { const float *w, *b; };
    int pi = 7;
    auto cpg = [&]() { CP p{(const float*)d_in[pi], (const float*)d_in[pi+1], (const float*)d_in[pi+2]}; pi += 3; return p; };
    auto pcg = [&]() { PC p{(const float*)d_in[pi], (const float*)d_in[pi+1]}; pi += 2; return p; };

    // params in make_params insertion order (nested dict insertion order)
    CP cb0 = cpg(), cb1 = cpg();
    CP c2cv1 = cpg(), c2cv2 = cpg(), c2m0a = cpg(), c2m0b = cpg();
    CP cb3 = cpg();
    CP c4cv1 = cpg(), c4cv2 = cpg(), c4m0a = cpg(), c4m0b = cpg(), c4m1a = cpg(), c4m1b = cpg();
    CP cb5 = cpg(), dn0 = cpg(), dn1 = cpg();
    CP le0 = cpg(), le1 = cpg();
    CP l2cv1 = cpg(), l2cv2 = cpg(), l2m0a = cpg(), l2m0b = cpg();
    CP l3cv1 = cpg(), l3cv2 = cpg();
    CP pcv1 = cpg(), pqkv = cpg(), pproj = cpg(), ppe = cpg(), pffn1 = cpg(), pffn2 = cpg(), pcv2 = cpg();
    PC clq = pcg(), clk = pcg(), clv = pcg();
    PC lcq = pcg(), lck = pcg(), lcv = pcg();
    CP fusep = cpg(), bh0 = cpg();
    CP b1cv1 = cpg(), b1cv2 = cpg(), b1m0a = cpg(), b1m0b = cpg(), b1m1a = cpg(), b1m1b = cpg();
    CP b2cv1 = cpg(), b2cv2 = cpg();
    CP bh3 = cpg();
    PC clsp = pcg(), regp = pcg(), ioup = pcg();

    // workspace bump allocator
    char* base = (char*)d_ws; size_t off = 0;
    auto alloc = [&](size_t nf) { float* p = (float*)(base + off); off += ((nf * 4 + 255) / 256) * 256; return p; };

    float* t0   = alloc((size_t)6 * 32 * 64 * 176);
    float* t1   = alloc((size_t)6 * 64 * 32 * 88);
    float* cat2 = alloc((size_t)6 * 96 * 32 * 88);
    float* tmp2 = alloc((size_t)6 * 32 * 32 * 88);
    float* x2   = alloc((size_t)6 * 64 * 32 * 88);
    float* t3   = alloc((size_t)6 * 128 * 16 * 44);
    float* cat4 = alloc((size_t)6 * 256 * 16 * 44);
    float* tmp4 = alloc((size_t)6 * 64 * 16 * 44);
    float* x4   = alloc((size_t)6 * 128 * 16 * 44);
    float* t5   = alloc((size_t)6 * 64 * 8 * 22);
    float* d0b  = alloc((size_t)6 * 64 * 8 * 22);
    float* d1b  = alloc((size_t)6 * 41 * 8 * 22);
    float* gp   = alloc(6 * 24);
    float* vol  = alloc((size_t)64 * 6400);
    float* l0b  = alloc((size_t)32 * 160 * 160);
    float* l1b  = alloc((size_t)64 * 6400);
    float* catL2 = alloc((size_t)96 * 6400);
    float* tmpL2 = alloc((size_t)32 * 6400);
    float* l2b  = alloc((size_t)64 * 6400);
    float* catL3 = alloc((size_t)128 * 6400);
    float* l3b  = alloc((size_t)64 * 6400);
    float* catP = alloc((size_t)64 * 6400);
    float* qkv  = alloc((size_t)64 * 6400);
    float* ob   = alloc((size_t)32 * 6400);
    float* o2   = alloc((size_t)32 * 6400);
    float* bb2  = alloc((size_t)32 * 6400);
    float* ftb  = alloc((size_t)64 * 6400);
    float* l4b  = alloc((size_t)64 * 6400);
    float* qc   = alloc((size_t)64 * 6400);
    float* kc   = alloc((size_t)64 * 6400);
    float* vc   = alloc((size_t)64 * 6400);
    float* ql   = alloc((size_t)64 * 6400);
    float* kl   = alloc((size_t)64 * 6400);
    float* vl   = alloc((size_t)64 * 6400);
    float* catF = alloc((size_t)128 * 6400);
    float* fu   = alloc((size_t)128 * 6400);
    float* h0   = alloc((size_t)128 * 6400);
    float* catB1 = alloc((size_t)512 * 6400);
    float* tmpB1 = alloc((size_t)128 * 6400);
    float* hb1  = alloc((size_t)256 * 6400);
    float* catB2 = alloc((size_t)512 * 6400);
    float* hb2  = alloc((size_t)256 * 6400);
    float* hb3  = alloc((size_t)384 * 6400);

    auto conv = [&](const float* in, CP p, float* out, int Bn, int Cin, int Cout,
                    int H, int W, int KH, int KW, int st, int pad,
                    int CinTot, int icoff, int CoutTot, int ocoff,
                    const float* res, int resCt, int resoff, int act) {
        int OH = (H + 2 * pad - KH) / st + 1;
        int OW = (W + 2 * pad - KW) / st + 1;
        dim3 g((OH * OW + 63) / 64, (Cout + 15) / 16, Bn);
        conv_wmma<<<g, 32, 0, stream>>>(in, p.w, p.s, p.b, res, out,
                                        Cin, Cout, H, W, OH, OW, KH, KW, st, pad,
                                        CinTot, icoff, CoutTot, ocoff, resCt, resoff, act);
    };
    auto pconv = [&](const float* in, PC p, float* out, int Cin, int Cout,
                     int CinTot, int icoff, int CoutTot, int ocoff) {
        CP q{p.w, nullptr, p.b};
        conv(in, q, out, 1, Cin, Cout, 80, 80, 1, 1, 1, 0,
             CinTot, icoff, CoutTot, ocoff, nullptr, 0, 0, 0);
    };

    // ---------------- camera backbone (BN=6) ----------------
    conv(cam, cb0, t0, 6, 3, 32, 128, 352, 7, 7, 2, 3, 3, 0, 32, 0, nullptr, 0, 0, 1);
    conv(t0, cb1, t1, 6, 32, 64, 64, 176, 3, 3, 2, 1, 32, 0, 64, 0, nullptr, 0, 0, 1);
    // cb2 = C2f(64,64,1)
    conv(t1, c2cv1, cat2, 6, 64, 64, 32, 88, 1, 1, 1, 0, 64, 0, 96, 0, nullptr, 0, 0, 1);
    conv(cat2, c2m0a, tmp2, 6, 32, 32, 32, 88, 3, 3, 1, 1, 96, 32, 32, 0, nullptr, 0, 0, 1);
    conv(tmp2, c2m0b, cat2, 6, 32, 32, 32, 88, 3, 3, 1, 1, 32, 0, 96, 64, nullptr, 0, 0, 1);
    conv(cat2, c2cv2, x2, 6, 96, 64, 32, 88, 1, 1, 1, 0, 96, 0, 64, 0, nullptr, 0, 0, 1);
    conv(x2, cb3, t3, 6, 64, 128, 32, 88, 3, 3, 2, 1, 64, 0, 128, 0, nullptr, 0, 0, 1);
    // cb4 = C2f(128,128,2)
    conv(t3, c4cv1, cat4, 6, 128, 128, 16, 44, 1, 1, 1, 0, 128, 0, 256, 0, nullptr, 0, 0, 1);
    conv(cat4, c4m0a, tmp4, 6, 64, 64, 16, 44, 3, 3, 1, 1, 256, 64, 64, 0, nullptr, 0, 0, 1);
    conv(tmp4, c4m0b, cat4, 6, 64, 64, 16, 44, 3, 3, 1, 1, 64, 0, 256, 128, nullptr, 0, 0, 1);
    conv(cat4, c4m1a, tmp4, 6, 64, 64, 16, 44, 3, 3, 1, 1, 256, 128, 64, 0, nullptr, 0, 0, 1);
    conv(tmp4, c4m1b, cat4, 6, 64, 64, 16, 44, 3, 3, 1, 1, 64, 0, 256, 192, nullptr, 0, 0, 1);
    conv(cat4, c4cv2, x4, 6, 256, 128, 16, 44, 1, 1, 1, 0, 256, 0, 128, 0, nullptr, 0, 0, 1);
    conv(x4, cb5, t5, 6, 128, 64, 16, 44, 3, 3, 2, 1, 128, 0, 64, 0, nullptr, 0, 0, 1);
    // depth net + softmax
    conv(t5, dn0, d0b, 6, 64, 64, 8, 22, 3, 3, 1, 1, 64, 0, 64, 0, nullptr, 0, 0, 1);
    conv(d0b, dn1, d1b, 6, 64, 41, 8, 22, 1, 1, 1, 0, 64, 0, 41, 0, nullptr, 0, 0, 1);
    softmax_ch<<<(6 * 176 + 255) / 256, 256, 0, stream>>>(d1b, 41, 176, 6);
    // lift-splat voxel pool
    fill_zero<<<(64 * 6400 + 255) / 256, 256, 0, stream>>>(vol, 64 * 6400);
    geom_prep<<<1, 32, 0, stream>>>(rots, trans, intr, prots, ptrans, gp);
    lift_voxel<<<(6 * 41 * 8 * 22 + 255) / 256, 256, 0, stream>>>(t5, d1b, gp, vol);

    // ---------------- lidar encoder (B=1) ----------------
    conv(lidar, le0, l0b, 1, 18, 32, 160, 160, 3, 3, 1, 1, 18, 0, 32, 0, nullptr, 0, 0, 1);
    conv(l0b, le1, l1b, 1, 32, 64, 160, 160, 3, 3, 2, 1, 32, 0, 64, 0, nullptr, 0, 0, 1);
    conv(l1b, l2cv1, catL2, 1, 64, 64, 80, 80, 1, 1, 1, 0, 64, 0, 96, 0, nullptr, 0, 0, 1);
    conv(catL2, l2m0a, tmpL2, 1, 32, 32, 80, 80, 3, 3, 1, 1, 96, 32, 32, 0, nullptr, 0, 0, 1);
    conv(tmpL2, l2m0b, catL2, 1, 32, 32, 80, 80, 3, 3, 1, 1, 32, 0, 96, 64, nullptr, 0, 0, 1);
    conv(catL2, l2cv2, l2b, 1, 96, 64, 80, 80, 1, 1, 1, 0, 96, 0, 64, 0, nullptr, 0, 0, 1);
    // SPPF
    conv(l2b, l3cv1, catL3, 1, 64, 32, 80, 80, 1, 1, 1, 0, 64, 0, 128, 0, nullptr, 0, 0, 1);
    maxpool5<<<(32 * 6400 + 255) / 256, 256, 0, stream>>>(catL3, catL3, 32, 80, 80, 0, 32);
    maxpool5<<<(32 * 6400 + 255) / 256, 256, 0, stream>>>(catL3, catL3, 32, 80, 80, 32, 64);
    maxpool5<<<(32 * 6400 + 255) / 256, 256, 0, stream>>>(catL3, catL3, 32, 80, 80, 64, 96);
    conv(catL3, l3cv2, l3b, 1, 128, 64, 80, 80, 1, 1, 1, 0, 128, 0, 64, 0, nullptr, 0, 0, 1);
    // PSA
    conv(l3b, pcv1, catP, 1, 64, 64, 80, 80, 1, 1, 1, 0, 64, 0, 64, 0, nullptr, 0, 0, 1);
    conv(catP, pqkv, qkv, 1, 32, 64, 80, 80, 1, 1, 1, 0, 64, 32, 64, 0, nullptr, 0, 0, 0);
    attn_fused<1, 2><<<400, 32, 0, stream>>>(qkv, 0, qkv, 16, qkv, 32, ob, 0,
                                             16, 6400, 6400, 0.25f);
    dwconv3<<<(32 * 6400 + 255) / 256, 256, 0, stream>>>(qkv, ppe.w, ppe.s, ppe.b,
                                                         ob, o2, 32, 80, 80, 32, 0, 0, 0);
    conv(o2, pproj, bb2, 1, 32, 32, 80, 80, 1, 1, 1, 0, 32, 0, 32, 0, catP, 64, 32, 0);
    conv(bb2, pffn1, ftb, 1, 32, 64, 80, 80, 1, 1, 1, 0, 32, 0, 64, 0, nullptr, 0, 0, 1);
    conv(ftb, pffn2, catP, 1, 64, 32, 80, 80, 1, 1, 1, 0, 64, 0, 64, 32, bb2, 32, 0, 0);
    conv(catP, pcv2, l4b, 1, 64, 64, 80, 80, 1, 1, 1, 0, 64, 0, 64, 0, nullptr, 0, 0, 1);

    // ---------------- cross-attention fusion ----------------
    pconv(vol, clq, qc, 64, 64, 64, 0, 64, 0);
    pconv(l4b, clk, kc, 64, 64, 64, 0, 64, 0);
    pconv(l4b, clv, vc, 64, 64, 64, 0, 64, 0);
    attn_fused<2, 4><<<400, 32, 0, stream>>>(qc, 0, kc, 0, vc, 0, catF, 0,
                                             64, 6400, 6400, 0.125f);
    pconv(l4b, lcq, ql, 64, 64, 64, 0, 64, 0);
    pconv(vol, lck, kl, 64, 64, 64, 0, 64, 0);
    pconv(vol, lcv, vl, 64, 64, 64, 0, 64, 0);
    attn_fused<2, 4><<<400, 32, 0, stream>>>(ql, 0, kl, 0, vl, 0, catF, 64,
                                             64, 6400, 6400, 0.125f);
    conv(catF, fusep, fu, 1, 128, 128, 80, 80, 1, 1, 1, 0, 128, 0, 128, 0, nullptr, 0, 0, 1);

    // ---------------- BEV head ----------------
    conv(fu, bh0, h0, 1, 128, 128, 80, 80, 1, 1, 1, 0, 128, 0, 128, 0, nullptr, 0, 0, 1);
    conv(h0, b1cv1, catB1, 1, 128, 256, 80, 80, 1, 1, 1, 0, 128, 0, 512, 0, nullptr, 0, 0, 1);
    conv(catB1, b1m0a, tmpB1, 1, 128, 128, 80, 80, 3, 3, 1, 1, 512, 128, 128, 0, nullptr, 0, 0, 1);
    conv(tmpB1, b1m0b, catB1, 1, 128, 128, 80, 80, 3, 3, 1, 1, 128, 0, 512, 256, nullptr, 0, 0, 1);
    conv(catB1, b1m1a, tmpB1, 1, 128, 128, 80, 80, 3, 3, 1, 1, 512, 256, 128, 0, nullptr, 0, 0, 1);
    conv(tmpB1, b1m1b, catB1, 1, 128, 128, 80, 80, 3, 3, 1, 1, 128, 0, 512, 384, nullptr, 0, 0, 1);
    conv(catB1, b1cv2, hb1, 1, 512, 256, 80, 80, 1, 1, 1, 0, 512, 0, 256, 0, nullptr, 0, 0, 1);
    conv(hb1, b2cv1, catB2, 1, 256, 128, 80, 80, 1, 1, 1, 0, 256, 0, 512, 0, nullptr, 0, 0, 1);
    maxpool5<<<(128 * 6400 + 255) / 256, 256, 0, stream>>>(catB2, catB2, 128, 80, 80, 0, 128);
    maxpool5<<<(128 * 6400 + 255) / 256, 256, 0, stream>>>(catB2, catB2, 128, 80, 80, 128, 256);
    maxpool5<<<(128 * 6400 + 255) / 256, 256, 0, stream>>>(catB2, catB2, 128, 80, 80, 256, 384);
    conv(catB2, b2cv2, hb2, 1, 512, 256, 80, 80, 1, 1, 1, 0, 512, 0, 256, 0, nullptr, 0, 0, 1);
    conv(hb2, bh3, hb3, 1, 256, 384, 80, 80, 1, 1, 1, 0, 256, 0, 384, 0, nullptr, 0, 0, 1);

    float* outf = (float*)d_out;
    pconv(hb3, clsp, outf,               384, 10, 384, 0, 10, 0);
    pconv(hb3, regp, outf + 10 * 6400,   384,  9, 384, 0,  9, 0);
    pconv(hb3, ioup, outf + 19 * 6400,   384,  1, 384, 0,  1, 0);
}
